// LGSBlock_85452669321526
// MI455X (gfx1250) — matrix operand order repaired and checked by reading, not compiled, not run on hardware
//
#include <hip/hip_runtime.h>

typedef __attribute__((ext_vector_type(16))) _Float16 v16h;
typedef __attribute__((ext_vector_type(8)))  _Float16 v8h;
typedef __attribute__((ext_vector_type(8)))  float    v8f;
typedef __attribute__((ext_vector_type(4)))  int      v4i32;

#define TM 64
#define TN 64
#define TK 32
#define LDSROW 40                    // 32 halfs + 8 pad -> 80B rows, 16B aligned
#define BOFF   (64 * LDSROW)         // B tile offset inside one LDS buffer (halfs)
#define BUFSZ  (2 * 64 * LDSROW)     // one double-buffer slot A+B (halfs)

#if __has_builtin(__builtin_amdgcn_global_load_async_to_lds_b128)
#define HAVE_ASYNC_LDS 1
#else
#define HAVE_ASYNC_LDS 0
#endif

__device__ __forceinline__ void cp16_g2l(const void* g, void* l)
{
#if HAVE_ASYNC_LDS
    __builtin_amdgcn_global_load_async_to_lds_b128(
        (__attribute__((address_space(1))) v4i32*)(g),
        (__attribute__((address_space(3))) v4i32*)(l), 0, 0);
#else
    *(int4*)l = *(const int4*)g;
#endif
}

__device__ __forceinline__ void async_wait_all()
{
#if HAVE_ASYNC_LDS
#if __has_builtin(__builtin_amdgcn_s_wait_asynccnt)
    __builtin_amdgcn_s_wait_asynccnt(0);
#else
    asm volatile("s_wait_asynccnt 0x0" ::: "memory");
#endif
#endif
}

// ---------------------------------------------------------------------------
// f16 tiled GEMM with double-buffered async LDS staging.
//   C[M,N] = epi(A[M,K] @ Bt[N,K]^T + bias (+ resid))
// A row-major f16 (lda), Bt row-major N x K f16 (ldb = K stride).
// Outputs: optional f32 C and/or f16 C16 (same ldc).
// epilogue: 0=none, 1=exact GELU, 2=sigmoid, 3=residual add
// ---------------------------------------------------------------------------
__global__ __launch_bounds__(128) void wmma_gemm_kernel(
    const _Float16* __restrict__ A, int lda,
    const _Float16* __restrict__ Bt, int ldb,
    const float* __restrict__ bias,
    const float* __restrict__ resid, int ldr,
    float* __restrict__ C, _Float16* __restrict__ C16, int ldc,
    int M, int N, int K, int epilogue)
{
    __shared__ __align__(16) _Float16 sm[2][BUFSZ];

    const int tid  = threadIdx.x;
    const int wave = tid >> 5;
    const int lane = tid & 31;
    const int ln16 = lane & 15;
    const int hi   = (lane >= 16) ? 1 : 0;

    const int m0 = blockIdx.y * TM;
    const int n0 = blockIdx.x * TN;

    v8f acc[4];
    #pragma unroll
    for (int i = 0; i < 4; ++i)
        acc[i] = (v8f){0.f,0.f,0.f,0.f,0.f,0.f,0.f,0.f};

    const int nk = K / TK;

    // stage tile kt into LDS buffer buf (issue only; completion via asynccnt)
    auto stage = [&](int buf, int kt) {
        const int k0 = kt * TK;
        // A tile: 64 rows x 4 chunks of 8 halfs
        #pragma unroll
        for (int i = tid; i < 256; i += 128) {
            int r = i >> 2, ch = i & 3;
            const _Float16* g = A + (long)(m0 + r) * lda + k0 + 8 * ch;
            cp16_g2l(g, &sm[buf][r * LDSROW + 8 * ch]);
        }
        // B tile (pre-transposed): 64 rows (n) x 4 chunks of 8 halfs (k)
        #pragma unroll
        for (int i = tid; i < 256; i += 128) {
            int r = i >> 2, ch = i & 3;
            if (n0 + r < N) {
                const _Float16* g = Bt + (long)(n0 + r) * ldb + k0 + 8 * ch;
                cp16_g2l(g, &sm[buf][BOFF + r * LDSROW + 8 * ch]);
            }
        }
    };

    stage(0, 0);
    int cur = 0;
    for (int kt = 0; kt < nk; ++kt) {
        async_wait_all();
        __syncthreads();
        if (kt + 1 < nk) stage(cur ^ 1, kt + 1);

        const v8h* base = (const v8h*)&sm[cur][0];
        // B fragment: lane column = wave*16+ln16; K = 16*hi + s  -> chunks 2hi, 2hi+1
        const int nrow = wave * 16 + ln16;
        v8h b0 = base[(BOFF >> 3) + nrow * 5 + 2 * hi];
        v8h b1 = base[(BOFF >> 3) + nrow * 5 + 2 * hi + 1];
        v16h bf = __builtin_shufflevector(b0, b1, 0,1,2,3,4,5,6,7,8,9,10,11,12,13,14,15);

        #pragma unroll
        for (int ms = 0; ms < 4; ++ms) {
            // A fragment: row = ms*16+ln16; K = 16*hi + s -> chunks 2hi, 2hi+1
            const int arow = ms * 16 + ln16;
            v8h a0 = base[arow * 5 + 2 * hi];
            v8h a1 = base[arow * 5 + 2 * hi + 1];
            v16h af = __builtin_shufflevector(a0, a1, 0,1,2,3,4,5,6,7,8,9,10,11,12,13,14,15);
            acc[ms] = __builtin_amdgcn_wmma_f32_16x16x32_f16(
                false, af, false, bf, (short)0, acc[ms], false, false);
        }
        cur ^= 1;
    }

    // store: C/D layout -> lane: N = ln16, M = r + 8*hi within 16x16 tile
    const int gn = n0 + wave * 16 + ln16;
    if (gn < N) {
        float bv = bias ? bias[gn] : 0.f;
        #pragma unroll
        for (int ms = 0; ms < 4; ++ms) {
            #pragma unroll
            for (int r = 0; r < 8; ++r) {
                int gm = m0 + ms * 16 + r + 8 * hi;
                if (gm < M) {
                    float v = acc[ms][r] + bv;
                    if (epilogue == 1) {
                        v = 0.5f * v * (1.0f + erff(v * 0.7071067811865475f));
                    } else if (epilogue == 2) {
                        v = 1.0f / (1.0f + expf(-v));
                    } else if (epilogue == 3) {
                        v += resid[(long)gm * ldr + gn];
                    }
                    if (C)   C[(long)gm * ldc + gn]   = v;
                    if (C16) C16[(long)gm * ldc + gn] = (_Float16)v;
                }
            }
        }
    }
}

// ---------------------------------------------------------------------------
// Weight convert + transpose: W[K,N] f32 -> Wt[N,K] f16 (32x32 LDS tiles)
// ---------------------------------------------------------------------------
__global__ __launch_bounds__(256) void convert_transpose_kernel(
    const float* __restrict__ W, _Float16* __restrict__ Wt, int K, int N)
{
    __shared__ float tile[32][33];
    int k0 = blockIdx.y * 32, n0 = blockIdx.x * 32;
    int tx = threadIdx.x & 31, ty = threadIdx.x >> 5;
    #pragma unroll
    for (int i = ty; i < 32; i += 8) {
        int k = k0 + i, n = n0 + tx;
        tile[i][tx] = (k < K && n < N) ? W[(long)k * N + n] : 0.f;
    }
    __syncthreads();
    #pragma unroll
    for (int i = ty; i < 32; i += 8) {
        int n = n0 + i, k = k0 + tx;
        if (n < N && k < K) Wt[(long)n * K + k] = (_Float16)tile[tx][i];
    }
}

// ---------------------------------------------------------------------------
// LayerNorm over last dim C (block per token); optional f32 / f16 outputs
// ---------------------------------------------------------------------------
__global__ __launch_bounds__(128) void ln_kernel(
    const float* __restrict__ x, const float* __restrict__ w,
    const float* __restrict__ b, float* __restrict__ out32,
    _Float16* __restrict__ out16, int C)
{
    long row = blockIdx.x;
    const float* xr = x + row * C;
    __shared__ float red[128];
    int tid = threadIdx.x;

    float s = 0.f;
    for (int c = tid; c < C; c += 128) s += xr[c];
    red[tid] = s; __syncthreads();
    for (int off = 64; off > 0; off >>= 1) { if (tid < off) red[tid] += red[tid + off]; __syncthreads(); }
    float mu = red[0] / C; __syncthreads();

    float v = 0.f;
    for (int c = tid; c < C; c += 128) { float d = xr[c] - mu; v += d * d; }
    red[tid] = v; __syncthreads();
    for (int off = 64; off > 0; off >>= 1) { if (tid < off) red[tid] += red[tid + off]; __syncthreads(); }
    float rstd = rsqrtf(red[0] / C + 1e-5f); __syncthreads();

    for (int c = tid; c < C; c += 128) {
        float o = (xr[c] - mu) * rstd * w[c] + b[c];
        if (out32) out32[row * C + c] = o;
        if (out16) out16[row * C + c] = (_Float16)o;
    }
}

// ---------------------------------------------------------------------------
// Causal depthwise conv1d (kernel 4) + SiLU on xBC slice of zxbcdt
// ---------------------------------------------------------------------------
__global__ __launch_bounds__(256) void conv_silu_kernel(
    const float* __restrict__ zx, const float* __restrict__ cw,
    const float* __restrict__ cb, float* __restrict__ out,
    int L, int ldz, int cd, int coloff, long total)
{
    long i = (long)blockIdx.x * 256 + threadIdx.x;
    if (i >= total) return;
    int c = (int)(i % cd);
    long bl = i / cd;
    int l = (int)(bl % L);
    long bbase = (bl - l) * ldz;

    float acc = cb[c];
    #pragma unroll
    for (int j = 0; j < 4; ++j) {
        int t = l - 3 + j;
        if (t >= 0) acc += cw[c * 4 + j] * zx[bbase + (long)t * ldz + coloff + c];
    }
    out[bl * cd + c] = acc / (1.f + expf(-acc));
}

// ---------------------------------------------------------------------------
// dt = softplus(dt_raw + dt_bias); dA = exp(dt * -exp(A_log))
// ---------------------------------------------------------------------------
__global__ __launch_bounds__(256) void dt_kernel(
    const float* __restrict__ zx, const float* __restrict__ dt_bias,
    const float* __restrict__ A_log, float* __restrict__ dtv,
    float* __restrict__ dAv, int ldz, int coloff, int nh, long total)
{
    long i = (long)blockIdx.x * 256 + threadIdx.x;
    if (i >= total) return;
    int h = (int)(i % nh);
    long row = i / nh;
    float t = zx[row * ldz + coloff + h] + dt_bias[h];
    float sp = (t > 20.f) ? t : log1pf(expf(t));
    dtv[i] = sp;
    dAv[i] = expf(sp * (-expf(A_log[h])));
}

// ---------------------------------------------------------------------------
// Mamba2 selective scan: block per (b,head), thread per headdim p.
// ---------------------------------------------------------------------------
__global__ __launch_bounds__(64) void scan_kernel(
    const float* __restrict__ xbc, const float* __restrict__ dtv,
    const float* __restrict__ dAv, const float* __restrict__ Dp,
    float* __restrict__ ybuf, int L)
{
    int bh = blockIdx.x;
    int b = bh / 12, h = bh % 12;
    int p = threadIdx.x;
    __shared__ float Bs[64], Cs[64];

    float hst[64];
    #pragma unroll
    for (int n = 0; n < 64; ++n) hst[n] = 0.f;
    float Dh = Dp[h];

    for (int t = 0; t < L; ++t) {
        long row = (long)b * L + t;
        Bs[p] = xbc[row * 896 + 768 + p];
        Cs[p] = xbc[row * 896 + 832 + p];
        __syncthreads();
        float dA = dAv[row * 12 + h];
        float dt = dtv[row * 12 + h];
        float xv = xbc[row * 896 + h * 64 + p];
        float a = dt * xv;
        float y = 0.f;
        #pragma unroll
        for (int n = 0; n < 64; ++n) {
            hst[n] = hst[n] * dA + a * Bs[n];
            y += hst[n] * Cs[n];
        }
        ybuf[row * 768 + h * 64 + p] = y + Dh * xv;
        __syncthreads();
    }
}

// ---------------------------------------------------------------------------
// g = y * silu(z); g *= rsqrt(mean(g^2)+eps) * mnorm_w  -> f16 out
// ---------------------------------------------------------------------------
__global__ __launch_bounds__(256) void gate_rms_kernel(
    const float* __restrict__ y, const float* __restrict__ zx,
    const float* __restrict__ mw, _Float16* __restrict__ out16,
    int di, int ldz)
{
    long row = blockIdx.x;
    int tid = threadIdx.x;
    __shared__ float red[256];

    float gv[3]; float ss = 0.f;
    #pragma unroll
    for (int i = 0; i < 3; ++i) {
        int c = tid + i * 256;
        float z = zx[row * ldz + c];
        float g = y[row * di + c] * (z / (1.f + expf(-z)));
        gv[i] = g; ss += g * g;
    }
    red[tid] = ss; __syncthreads();
    for (int off = 128; off > 0; off >>= 1) { if (tid < off) red[tid] += red[tid + off]; __syncthreads(); }
    float r = rsqrtf(red[0] / di + 1e-5f);
    #pragma unroll
    for (int i = 0; i < 3; ++i) {
        int c = tid + i * 256;
        out16[row * di + c] = (_Float16)(gv[i] * r * mw[c]);
    }
}

// ---------------------------------------------------------------------------
// Gated 7x7 window attention: block per (window, head); f16 output
// ---------------------------------------------------------------------------
__global__ __launch_bounds__(64) void attn_kernel(
    const float* __restrict__ qkv, const float* __restrict__ gate,
    _Float16* __restrict__ xa16)
{
    const int L = 56 * 56;
    int head = blockIdx.x % 12;
    int w    = blockIdx.x / 12;
    int b = w >> 6; int wrem = w & 63;
    int hb = wrem >> 3, wb = wrem & 7;

    __shared__ float Kd[49][33];
    __shared__ float Vd[49][33];

    for (int i = threadIdx.x; i < 49 * 32; i += 64) {
        int m = i >> 5, d = i & 31;
        int r = m / 7, cc = m % 7;
        int l = (hb * 7 + r) * 56 + (wb * 7 + cc);
        long base = ((long)b * L + l) * 1152 + head * 32 + d;
        Kd[m][d] = qkv[base + 384];
        Vd[m][d] = qkv[base + 768];
    }
    __syncthreads();

    int n = threadIdx.x;
    if (n < 49) {
        int r = n / 7, cc = n % 7;
        int l = (hb * 7 + r) * 56 + (wb * 7 + cc);
        long qbase = ((long)b * L + l) * 1152 + head * 32;
        const float scale = 0.17677669529663687f;
        float q[32];
        #pragma unroll
        for (int d = 0; d < 32; ++d) q[d] = qkv[qbase + d] * scale;

        float sc[49]; float mx = -1e30f;
        #pragma unroll
        for (int m = 0; m < 49; ++m) {
            float s = 0.f;
            #pragma unroll
            for (int d = 0; d < 32; ++d) s += q[d] * Kd[m][d];
            sc[m] = s; mx = fmaxf(mx, s);
        }
        float se = 0.f;
        #pragma unroll
        for (int m = 0; m < 49; ++m) { sc[m] = expf(sc[m] - mx); se += sc[m]; }
        float inv = 1.f / se;
        float gv = gate[((long)b * L + l) * 12 + head];
        #pragma unroll
        for (int d = 0; d < 32; ++d) {
            float o = 0.f;
            #pragma unroll
            for (int m = 0; m < 49; ++m) o += sc[m] * Vd[m][d];
            xa16[((long)b * L + l) * 384 + head * 32 + d] = (_Float16)(o * inv * gv);
        }
    }
}

// ---------------------------------------------------------------------------
static inline void launch_gemm(const _Float16* A, int lda, const _Float16* Bt, int ldb,
                               const float* bias, const float* resid, int ldr,
                               float* C, _Float16* C16, int ldc,
                               int M, int N, int K, int epi, hipStream_t stream)
{
    dim3 grid((N + TN - 1) / TN, (M + TM - 1) / TM);
    wmma_gemm_kernel<<<grid, 128, 0, stream>>>(A, lda, Bt, ldb, bias, resid, ldr,
                                               C, C16, ldc, M, N, K, epi);
}

static inline void launch_convT(const float* W, _Float16* Wt, int K, int N,
                                hipStream_t stream)
{
    dim3 grid((N + 31) / 32, (K + 31) / 32);
    convert_transpose_kernel<<<grid, 256, 0, stream>>>(W, Wt, K, N);
}

extern "C" void kernel_launch(void* const* d_in, const int* in_sizes, int n_in,
                              void* d_out, int out_size, void* d_ws, size_t ws_size,
                              hipStream_t stream)
{
    const float* x      = (const float*)d_in[0];
    const float* n1w    = (const float*)d_in[1];
    const float* n1b    = (const float*)d_in[2];
    const float* W_in   = (const float*)d_in[3];
    const float* conv_w = (const float*)d_in[4];
    const float* conv_b = (const float*)d_in[5];
    const float* dt_bias= (const float*)d_in[6];
    const float* A_log  = (const float*)d_in[7];
    const float* Dp     = (const float*)d_in[8];
    const float* mnw    = (const float*)d_in[9];
    const float* W_out  = (const float*)d_in[10];
    const float* qkv_w  = (const float*)d_in[11];
    const float* qkv_b  = (const float*)d_in[12];
    const float* gate_w = (const float*)d_in[13];
    const float* gate_b = (const float*)d_in[14];
    const float* proj_w = (const float*)d_in[15];
    const float* proj_b = (const float*)d_in[16];
    const float* fus_w  = (const float*)d_in[17];
    const float* fus_b  = (const float*)d_in[18];
    const float* n2w    = (const float*)d_in[19];
    const float* n2b    = (const float*)d_in[20];
    const float* fc1_w  = (const float*)d_in[21];
    const float* fc1_b  = (const float*)d_in[22];
    const float* fc2_w  = (const float*)d_in[23];
    const float* fc2_b  = (const float*)d_in[24];
    float* out = (float*)d_out;

    const int  Bn = 8, C = 384, L = 56 * 56;
    const long BT = (long)Bn * L;                 // 25088
    const int  DIP = 1676;

    // ---- workspace bump allocator (256B aligned) ----
    char* p = (char*)d_ws;
    auto alloc = [&](size_t bytes) -> char* {
        char* r = p; p += (bytes + 255) & ~(size_t)255; return r;
    };
    // f16 transposed weights
    _Float16* w_in_t  = (_Float16*)alloc((size_t)DIP  * 384  * 2);
    _Float16* w_out_t = (_Float16*)alloc((size_t)384  * 768  * 2);
    _Float16* qkv_t   = (_Float16*)alloc((size_t)1152 * 384  * 2);
    _Float16* gate_t  = (_Float16*)alloc((size_t)12   * 384  * 2);
    _Float16* proj_t  = (_Float16*)alloc((size_t)384  * 384  * 2);
    _Float16* fus_t   = (_Float16*)alloc((size_t)384  * 768  * 2);
    _Float16* fc1_t   = (_Float16*)alloc((size_t)1536 * 384  * 2);
    _Float16* fc2_t   = (_Float16*)alloc((size_t)384  * 1536 * 2);
    // f16 activations
    _Float16* xn16    = (_Float16*)alloc((size_t)BT * 384  * 2);
    _Float16* xn2_16  = (_Float16*)alloc((size_t)BT * 384  * 2);
    _Float16* xa16    = (_Float16*)alloc((size_t)BT * 384  * 2);
    _Float16* ybuf16  = (_Float16*)alloc((size_t)BT * 768  * 2);
    _Float16* xcat16  = (_Float16*)alloc((size_t)BT * 768  * 2);
    _Float16* mid16   = (_Float16*)alloc((size_t)BT * 1536 * 2);
    // f32 buffers
    float* xres = (float*)alloc((size_t)BT * 384 * 4);
    float* ybuf = (float*)alloc((size_t)BT * 768 * 4);
    float* xbc  = (float*)alloc((size_t)BT * 896 * 4);
    float* dtv  = (float*)alloc((size_t)BT * 12  * 4);
    float* dAv  = (float*)alloc((size_t)BT * 12  * 4);
    float* zx   = (float*)alloc((size_t)BT * DIP * 4);
    float* qkv  = zx;                              // alias: zx dead before qkv GEMM
    float* gbuf = zx + BT * 1152;                  // inside dead zx tail

    // ---- 0. weight convert+transpose (f32 KxN -> f16 NxK) ----
    launch_convT(W_in,   w_in_t,  384,  DIP,  stream);
    launch_convT(W_out,  w_out_t, 768,  384,  stream);
    launch_convT(qkv_w,  qkv_t,   384,  1152, stream);
    launch_convT(gate_w, gate_t,  384,  12,   stream);
    launch_convT(proj_w, proj_t,  384,  384,  stream);
    launch_convT(fus_w,  fus_t,   768,  384,  stream);
    launch_convT(fc1_w,  fc1_t,   384,  1536, stream);
    launch_convT(fc2_w,  fc2_t,   1536, 384,  stream);

    // ---- 1. norm1 -> xn16 ----
    ln_kernel<<<(int)BT, 128, 0, stream>>>(x, n1w, n1b, nullptr, xn16, C);

    // ---- 2. zxbcdt = xn @ W_in ----
    launch_gemm(xn16, 384, w_in_t, 384, nullptr, nullptr, 0,
                zx, nullptr, DIP, (int)BT, DIP, 384, 0, stream);

    // ---- 3. causal depthwise conv + silu ----
    {
        long total = BT * 896;
        conv_silu_kernel<<<(int)((total + 255) / 256), 256, 0, stream>>>(
            zx, conv_w, conv_b, xbc, L, DIP, 896, 768, total);
    }
    // ---- 4. softplus(dt)+dA ----
    {
        long total = BT * 12;
        dt_kernel<<<(int)((total + 255) / 256), 256, 0, stream>>>(
            zx, dt_bias, A_log, dtv, dAv, DIP, 1664, 12, total);
    }
    // ---- 5. selective scan (+ D*x) ----
    scan_kernel<<<Bn * 12, 64, 0, stream>>>(xbc, dtv, dAv, Dp, ybuf, L);

    // ---- 6. silu(z)-gating + RMSNorm -> ybuf16 ----
    gate_rms_kernel<<<(int)BT, 256, 0, stream>>>(ybuf, zx, mnw, ybuf16, 768, DIP);

    // ---- 7. mamba out -> xcat16[:, 0:384] ----
    launch_gemm(ybuf16, 768, w_out_t, 768, nullptr, nullptr, 0,
                nullptr, xcat16, 768, (int)BT, 384, 768, 0, stream);

    // ---- 8. qkv (zx dead -> reuse) ----
    launch_gemm(xn16, 384, qkv_t, 384, qkv_b, nullptr, 0,
                qkv, nullptr, 1152, (int)BT, 1152, 384, 0, stream);

    // ---- 9. gate = sigmoid(xn @ gate_w + gate_b) ----
    launch_gemm(xn16, 384, gate_t, 384, gate_b, nullptr, 0,
                gbuf, nullptr, 12, (int)BT, 12, 384, 2, stream);

    // ---- 10. gated window attention -> xa16 ----
    attn_kernel<<<Bn * 64 * 12, 64, 0, stream>>>(qkv, gbuf, xa16);

    // ---- 11. attn proj -> xcat16[:, 384:768] ----
    launch_gemm(xa16, 384, proj_t, 384, proj_b, nullptr, 0,
                nullptr, xcat16 + 384, 768, (int)BT, 384, 384, 0, stream);

    // ---- 12. fusion + shortcut residual -> xres ----
    launch_gemm(xcat16, 768, fus_t, 768, fus_b, x, 384,
                xres, nullptr, 384, (int)BT, 384, 768, 3, stream);

    // ---- 13. norm2 -> xn2_16 ----
    ln_kernel<<<(int)BT, 128, 0, stream>>>(xres, n2w, n2b, nullptr, xn2_16, C);

    // ---- 14. fc1 + exact GELU -> mid16 ----
    launch_gemm(xn2_16, 384, fc1_t, 384, fc1_b, nullptr, 0,
                nullptr, mid16, 1536, (int)BT, 1536, 384, 1, stream);

    // ---- 15. fc2 + residual -> out ----
    launch_gemm(mid16, 1536, fc2_t, 1536, fc2_b, xres, 384,
                out, nullptr, 384, (int)BT, 384, 1536, 3, stream);
}
